// KANLinear_63462436766335
// MI455X (gfx1250) — matrix operand order repaired and checked by reading, not compiled, not run on hardware
//
#include <hip/hip_runtime.h>

typedef _Float16 v16h __attribute__((ext_vector_type(16)));
typedef _Float16 v8h  __attribute__((ext_vector_type(8)));
typedef float    v8f  __attribute__((ext_vector_type(8)));

union FragAB { v16h v; v8h h[2]; };

#define BATCH   16384
#define INF     512
#define OUTF    512
#define NB      8        // spline bases per input feature
#define TM      128      // workgroup M tile
#define TN      256      // workgroup N tile
#define LDA     40       // padded LDS row stride in halves (32 data + 8 pad, 80B = 5x16B)
#define P1      16       // phase-1 K-steps (silu part, K=512)
#define P2      128      // phase-2 K-steps (spline part, K=4096)

__device__ __forceinline__ float silu_f(float v) {
  return v / (1.0f + __expf(-v));
}

// 8 cubic B-spline bases on a *uniform* knot grid.  With u = (x - g0)/h:
//   left  = (u - t)/k ,  right = (t+k+1 - u)/k ,  b0[t] = (u>=t) - (u>=t+1)
// All coefficients are compile-time constants -> pure VALU, no LDS reads.
__device__ __forceinline__ void bases8u(float u, float outv[8]) {
  float c[12];
#pragma unroll
  for (int t = 0; t < 12; ++t) c[t] = (u >= (float)t) ? 1.0f : 0.0f;
  float b[11];
#pragma unroll
  for (int t = 0; t < 11; ++t) b[t] = c[t] - c[t + 1];
#pragma unroll
  for (int k = 1; k <= 3; ++k) {
    const float invk = (k == 1) ? 1.0f : (k == 2) ? 0.5f : (1.0f / 3.0f);
#pragma unroll
    for (int t = 0; t < 11 - k; ++t) {
      const float l = u - (float)t;
      const float r = (float)(t + k + 1) - u;
      b[t] = (l * b[t] + r * b[t + 1]) * invk;
    }
  }
#pragma unroll
  for (int t = 0; t < 8; ++t) outv[t] = b[t];
}

__global__ __launch_bounds__(256) void kan_wmma_kernel(
    const float* __restrict__ x,         // (16384, 512)
    const float* __restrict__ base_w,    // (512, 512)
    const float* __restrict__ spline_w,  // (512, 512, 8)
    const float* __restrict__ scaler,    // (512, 512)
    const float* __restrict__ grid,      // (512, 12) -- rows identical, uniform
    float* __restrict__ out)             // (16384, 512)
{
  // double-buffered tiles: 2 * (10 KB + 20 KB) = 60 KB of 320 KB WGP LDS
  __shared__ __align__(16) _Float16 As[2][TM * LDA];
  __shared__ __align__(16) _Float16 Bs[2][TN * LDA];

  const int tid = threadIdx.x;
  const int bm  = (int)blockIdx.x >> 1;   // 128 M-blocks
  const int bn  = (int)blockIdx.x & 1;    // 2   N-blocks
  const int m0g = bm * TM;
  const int n0g = bn * TN;

  // uniform-grid parameters (g0 = first knot, h = spacing)
  const float g0    = grid[0];
  const float inv_h = 1.0f / (grid[1] - grid[0]);

  // ---- lane / wave decomposition --------------------------------------
  const int lane = tid & 31;
  const int wid  = tid >> 5;     // 0..7
  const int wm   = wid & 1;      // 2 M wave-groups (64 rows each)
  const int wn   = wid >> 1;     // 4 N wave-groups (64 cols each)
  const int l16  = lane & 15;
  const int hi   = lane >> 4;    // half-wave select

  // staging assignment: A -> thread pair per row, B -> one row per thread
  const int arow  = tid >> 1;          // 0..127
  const int abase = (tid & 1) * 2;     // chunks {0,1} or {2,3}
  const int brow  = tid;               // 0..255

  const float* xrow = x + (size_t)(m0g + arow) * INF;

  const v8f vzero = {0.f, 0.f, 0.f, 0.f, 0.f, 0.f, 0.f, 0.f};
  v8f acc[4][4];
#pragma unroll
  for (int mi = 0; mi < 4; ++mi)
#pragma unroll
    for (int ni = 0; ni < 4; ++ni) acc[mi][ni] = vzero;

  // ---- phase-1 prep: silu(x) columns / base_weight ---------------------
  auto prep1 = [&](int step, v8h pA[2], v8h pB[4]) {
#pragma unroll
    for (int s = 0; s < 2; ++s) {
      const int k0 = step * 32 + (abase + s) * 8;
      const float4 v0 = *(const float4*)(xrow + k0);
      const float4 v1 = *(const float4*)(xrow + k0 + 4);
      float vals[8] = {silu_f(v0.x), silu_f(v0.y), silu_f(v0.z), silu_f(v0.w),
                       silu_f(v1.x), silu_f(v1.y), silu_f(v1.z), silu_f(v1.w)};
      v8h hh;
#pragma unroll
      for (int j = 0; j < 8; ++j) hh[j] = (_Float16)vals[j];
      pA[s] = hh;
    }
#pragma unroll
    for (int c = 0; c < 4; ++c) {
      const int k0 = step * 32 + c * 8;
      const float* wrow = base_w + (size_t)(n0g + brow) * INF + k0;
      const float4 v0 = *(const float4*)(wrow);
      const float4 v1 = *(const float4*)(wrow + 4);
      v8h hh;
      hh[0] = (_Float16)v0.x; hh[1] = (_Float16)v0.y;
      hh[2] = (_Float16)v0.z; hh[3] = (_Float16)v0.w;
      hh[4] = (_Float16)v1.x; hh[5] = (_Float16)v1.y;
      hh[6] = (_Float16)v1.z; hh[7] = (_Float16)v1.w;
      pB[c] = hh;
    }
  };

  // ---- phase-2 prep: spline bases / scaled spline weights --------------
  auto prep2 = [&](int step, v8h pA[2], v8h pB[4]) {
#pragma unroll
    for (int s = 0; s < 2; ++s) {
      const int i = step * 4 + (abase + s);
      const float u = (xrow[i] - g0) * inv_h;
      float vals[8];
      bases8u(u, vals);
      v8h hh;
#pragma unroll
      for (int j = 0; j < 8; ++j) hh[j] = (_Float16)vals[j];
      pA[s] = hh;
    }
#pragma unroll
    for (int c = 0; c < 4; ++c) {
      const int i = step * 4 + c;
      const float sc = scaler[(size_t)(n0g + brow) * INF + i];
      const float* wrow = spline_w + ((size_t)(n0g + brow) * INF + i) * NB;
      const float4 v0 = *(const float4*)(wrow);
      const float4 v1 = *(const float4*)(wrow + 4);
      v8h hh;
      hh[0] = (_Float16)(v0.x * sc); hh[1] = (_Float16)(v0.y * sc);
      hh[2] = (_Float16)(v0.z * sc); hh[3] = (_Float16)(v0.w * sc);
      hh[4] = (_Float16)(v1.x * sc); hh[5] = (_Float16)(v1.y * sc);
      hh[6] = (_Float16)(v1.z * sc); hh[7] = (_Float16)(v1.w * sc);
      pB[c] = hh;
    }
  };

  auto stage = [&](_Float16* as, _Float16* bs, const v8h pA[2], const v8h pB[4]) {
#pragma unroll
    for (int s = 0; s < 2; ++s)
      *(v8h*)&as[arow * LDA + (abase + s) * 8] = pA[s];
#pragma unroll
    for (int c = 0; c < 4; ++c)
      *(v8h*)&bs[brow * LDA + c * 8] = pB[c];
  };

  // ---- one K-step of WMMAs from LDS -----------------------------------
  auto compute = [&](const _Float16* as, const _Float16* bs) {
    FragAB bf[4], af[4];
#pragma unroll
    for (int ni = 0; ni < 4; ++ni) {
      // B fragment: lanes 0-15 hold K 0-15, lanes 16-31 hold K 16-31
      const v8h* rp = (const v8h*)&bs[(wn * 64 + ni * 16 + l16) * LDA];
      bf[ni].h[0] = rp[hi * 2];
      bf[ni].h[1] = rp[hi * 2 + 1];
    }
#pragma unroll
    for (int mi = 0; mi < 4; ++mi) {
      // A fragment: lanes 0-15 hold K 0-15, lanes 16-31 hold K 16-31
      const v8h* rp = (const v8h*)&as[(wm * 64 + mi * 16 + l16) * LDA];
      af[mi].h[0] = rp[hi * 2];
      af[mi].h[1] = rp[hi * 2 + 1];
    }
#pragma unroll
    for (int mi = 0; mi < 4; ++mi)
#pragma unroll
      for (int ni = 0; ni < 4; ++ni)
        acc[mi][ni] = __builtin_amdgcn_wmma_f32_16x16x32_f16(
            false, af[mi].v, false, bf[ni].v, (short)0, acc[mi][ni], false, false);
  };

  // ---- pipelined main loops (single barrier per K-step) ----------------
  v8h pA[2], pB[4];
  prep1(0, pA, pB);
  stage(As[0], Bs[0], pA, pB);
  __syncthreads();

  int par = 0;
  for (int t = 0; t < P1; ++t) {            // 16 silu steps
    v8h nA[2], nB[4];
    if (t + 1 < P1) prep1(t + 1, nA, nB);
    else            prep2(0, nA, nB);
    compute(As[par], Bs[par]);
    stage(As[par ^ 1], Bs[par ^ 1], nA, nB);
    par ^= 1;
    __syncthreads();
  }
  for (int t = 0; t < P2; ++t) {            // 128 spline steps (hot loop)
    v8h nA[2], nB[4];
    const bool more = (t + 1 < P2);
    if (more) prep2(t + 1, nA, nB);
    compute(As[par], Bs[par]);
    if (more) stage(As[par ^ 1], Bs[par ^ 1], nA, nB);
    par ^= 1;
    __syncthreads();
  }

  // ---- epilogue: f32 C layout (VGPR v: M=v / M=v+8 for half-waves) ----
#pragma unroll
  for (int mi = 0; mi < 4; ++mi) {
#pragma unroll
    for (int ni = 0; ni < 4; ++ni) {
      const int col = n0g + wn * 64 + ni * 16 + l16;
#pragma unroll
      for (int v = 0; v < 8; ++v) {
        const int row = m0g + wm * 64 + mi * 16 + v + hi * 8;
        out[(size_t)row * OUTF + col] = acc[mi][ni][v];
      }
    }
  }
}

extern "C" void kernel_launch(void* const* d_in, const int* in_sizes, int n_in,
                              void* d_out, int out_size, void* d_ws, size_t ws_size,
                              hipStream_t stream) {
  (void)in_sizes; (void)n_in; (void)d_ws; (void)ws_size; (void)out_size;
  const float* x        = (const float*)d_in[0];
  const float* base_w   = (const float*)d_in[1];
  const float* spline_w = (const float*)d_in[2];
  const float* scaler   = (const float*)d_in[3];
  const float* grid     = (const float*)d_in[4];
  float* out = (float*)d_out;

  dim3 grd((BATCH / TM) * (OUTF / TN));   // 128 * 2 = 256 workgroups
  dim3 blk(256);                          // 8 wave32 waves
  hipLaunchKernelGGL(kan_wmma_kernel, grd, blk, 0, stream,
                     x, base_w, spline_w, scaler, grid, out);
}
